// VectorQuantizer_68367289418176
// MI455X (gfx1250) — compile-verified, matrix operand order checked by reading, not run on hardware
//
#include <hip/hip_runtime.h>

typedef __attribute__((ext_vector_type(16))) __bf16     v16bf;
typedef __attribute__((ext_vector_type(8)))  float      v8f;
typedef __attribute__((ext_vector_type(4)))  float      f32x4;
typedef __attribute__((ext_vector_type(2)))  float      f32x2;
typedef __attribute__((ext_vector_type(8)))  unsigned short u16x8;

#define K_CODES   1024
#define D_DIM     64
#define N_TOK     131072            // 32 * 4096
#define EPAD      72                // padded row stride (bf16 elems): 144B, 16B aligned
#define QUANT_ELEMS 8388608         // N_TOK * D_DIM
#define LOSS_OFF    8388608
#define IDX_OFF     8388609
#define ROWS_PER_BLOCK 256
#define NBLOCKS   (N_TOK / ROWS_PER_BLOCK)   // 512

__device__ __forceinline__ unsigned short f2bf(float f) {
    unsigned u = __float_as_uint(f);
    unsigned r = u + 0x7FFFu + ((u >> 16) & 1u);   // round-to-nearest-even
    return (unsigned short)(r >> 16);
}

union FragA { unsigned short s[16]; v16bf v; };
union FragB { u16x8 h[2];           v16bf v; };

// build A-fragments (16x32 bf16 per-lane layout, ISA 7.12.2) for one X row;
// returns this lane's partial sum-of-squares (its 32 elements of the row)
__device__ __forceinline__ float buildA(const float* __restrict__ xrow, int hi,
                                        FragA& a0, FragA& a1) {
    const int off0 = hi * 8;
    float tmp[4][8];
    #pragma unroll
    for (int kcs = 0; kcs < 2; ++kcs) {
        #pragma unroll
        for (int grp = 0; grp < 2; ++grp) {
            const float* src = xrow + kcs * 32 + grp * 16 + off0;
            f32x4 q0 = *(const f32x4*)(src);
            f32x4 q1 = *(const f32x4*)(src + 4);
            #pragma unroll
            for (int i = 0; i < 4; ++i) {
                tmp[kcs * 2 + grp][i]     = q0[i];
                tmp[kcs * 2 + grp][4 + i] = q1[i];
            }
        }
    }
    float xs = 0.f;
    #pragma unroll
    for (int i = 0; i < 8; ++i) {
        xs += tmp[0][i] * tmp[0][i] + tmp[1][i] * tmp[1][i]
            + tmp[2][i] * tmp[2][i] + tmp[3][i] * tmp[3][i];
        a0.s[i]     = f2bf(tmp[0][i]);
        a0.s[8 + i] = f2bf(tmp[1][i]);
        a1.s[i]     = f2bf(tmp[2][i]);
        a1.s[8 + i] = f2bf(tmp[3][i]);
    }
    return xs;
}

__device__ __forceinline__ void loadB(const unsigned short* __restrict__ erow,
                                      int hi, FragB& b0, FragB& b1) {
    b0.h[0] = *(const u16x8*)(erow + hi * 8);
    b0.h[1] = *(const u16x8*)(erow + hi * 8 + 16);
    b1.h[0] = *(const u16x8*)(erow + 32 + hi * 8);
    b1.h[1] = *(const u16x8*)(erow + 32 + hi * 8 + 16);
}

// ---------------- prep: embedding f32 -> bf16 + per-code squared norms ---------
__global__ __launch_bounds__(256) void vq_prep(const float* __restrict__ emb,
                                               unsigned short* __restrict__ wsE,
                                               float* __restrict__ wsEsq) {
    int k = blockIdx.x * 256 + threadIdx.x;
    if (k >= K_CODES) return;
    const float* r = emb + (size_t)k * D_DIM;
    float s = 0.f;
    #pragma unroll
    for (int d = 0; d < D_DIM; ++d) {
        float f = r[d];
        s += f * f;
        wsE[(size_t)k * D_DIM + d] = f2bf(f);
    }
    wsEsq[k] = s;
}

// ---------------- main: WMMA distances + argmin + gather -----------------------
__global__ __launch_bounds__(256) void vq_main(const float* __restrict__ x,
                                               const float* __restrict__ emb,
                                               const unsigned short* __restrict__ wsE,
                                               const float* __restrict__ wsEsq,
                                               float* __restrict__ out,
                                               float* __restrict__ partials) {
    __shared__ unsigned short sE[K_CODES * EPAD];   // 147456 B, bf16 codebook (padded)
    __shared__ float sEsq[K_CODES];                 // 4 KB
    __shared__ float sXsq[8][32];
    __shared__ int   sBest[8][32];
    __shared__ float sLoss;

    const int tid  = threadIdx.x;
    const int wave = tid >> 5;
    const int lane = tid & 31;
    const int ln   = lane & 15;
    const int hi   = lane >> 4;
    const int rowBase  = blockIdx.x * ROWS_PER_BLOCK;
    const int waveBase = rowBase + wave * 32;       // 32 rows of X per wave

    if (tid == 0) sLoss = 0.f;

    // ---- async-copy bf16 codebook into LDS (padded rows), ASYNCcnt tracked ----
    #pragma unroll 4
    for (int it = 0; it < 32; ++it) {
        int chunk = tid + it * 256;                 // 8192 x 16B chunks
        int row = chunk >> 3;
        int col = chunk & 7;
        unsigned ldsOff = (unsigned)(unsigned long long)(&sE[row * EPAD + col * 8]);
        unsigned long long gaddr = (unsigned long long)(wsE + (size_t)chunk * 8);
        asm volatile("global_load_async_to_lds_b128 %0, %1, off"
                     :: "v"(ldsOff), "v"(gaddr) : "memory");
    }
    {   // esq: 256 threads x 16B = 4KB
        unsigned ldsOff = (unsigned)(unsigned long long)(&sEsq[tid * 4]);
        unsigned long long gaddr = (unsigned long long)(wsEsq + (size_t)tid * 4);
        asm volatile("global_load_async_to_lds_b128 %0, %1, off"
                     :: "v"(ldsOff), "v"(gaddr) : "memory");
    }
    asm volatile("s_wait_asynccnt 0" ::: "memory");
    __syncthreads();

    // ---- build A frags for two 16-row tiles (rows ln and 16+ln) + norms ----
    FragA a0, a1, a2, a3;
    float xs0 = buildA(x + (size_t)(waveBase + ln)      * D_DIM, hi, a0, a1);
    float xs1 = buildA(x + (size_t)(waveBase + 16 + ln) * D_DIM, hi, a2, a3);
    xs0 += __shfl_xor(xs0, 16);                     // lane m + lane m+16 = full row
    xs1 += __shfl_xor(xs1, 16);
    if (hi == 0) { sXsq[wave][ln] = xs0; sXsq[wave][16 + ln] = xs1; }

    // ---- sweep 1024 codes, two 16-code tiles per iteration:
    //      8 WMMAs into 4 independent chains; consumption of earlier chains
    //      provides the co-exec slots that cover each chain's D->VALU hazard ----
    float best[16];
    int   bidx[16];
    #pragma unroll
    for (int j = 0; j < 16; ++j) { best[j] = 3.4e38f; bidx[j] = 0; }

    for (int t = 0; t < 64; t += 2) {
        FragB b0, b1, b2, b3;
        loadB(&sE[((t + 0) * 16 + ln) * EPAD], hi, b0, b1);
        loadB(&sE[((t + 1) * 16 + ln) * EPAD], hi, b2, b3);

        v8f c00 = {}, c01 = {}, c10 = {}, c11 = {};
        c00 = __builtin_amdgcn_wmma_f32_16x16x32_bf16(false, a0.v, false, b0.v,
                                                      (short)0, c00, false, false);
        c01 = __builtin_amdgcn_wmma_f32_16x16x32_bf16(false, a2.v, false, b0.v,
                                                      (short)0, c01, false, false);
        c00 = __builtin_amdgcn_wmma_f32_16x16x32_bf16(false, a1.v, false, b1.v,
                                                      (short)0, c00, false, false);
        c01 = __builtin_amdgcn_wmma_f32_16x16x32_bf16(false, a3.v, false, b1.v,
                                                      (short)0, c01, false, false);
        c10 = __builtin_amdgcn_wmma_f32_16x16x32_bf16(false, a0.v, false, b2.v,
                                                      (short)0, c10, false, false);
        c11 = __builtin_amdgcn_wmma_f32_16x16x32_bf16(false, a2.v, false, b2.v,
                                                      (short)0, c11, false, false);
        c10 = __builtin_amdgcn_wmma_f32_16x16x32_bf16(false, a1.v, false, b3.v,
                                                      (short)0, c10, false, false);
        c11 = __builtin_amdgcn_wmma_f32_16x16x32_bf16(false, a3.v, false, b3.v,
                                                      (short)0, c11, false, false);

        const int   n0 = (t + 0) * 16 + ln;
        const int   n1 = (t + 1) * 16 + ln;
        const float e0 = sEsq[n0];
        const float e1 = sEsq[n1];
        #pragma unroll
        for (int j = 0; j < 8; ++j) {               // tile t: rows 0..15, 16..31
            float v0 = fmaf(-2.f, c00[j], e0);      // |e|^2 - 2 x.e  (|x|^2 const)
            float v1 = fmaf(-2.f, c01[j], e0);
            if (v0 < best[j])     { best[j]     = v0; bidx[j]     = n0; }
            if (v1 < best[8 + j]) { best[8 + j] = v1; bidx[8 + j] = n0; }
        }
        #pragma unroll
        for (int j = 0; j < 8; ++j) {               // tile t+1
            float v0 = fmaf(-2.f, c10[j], e1);
            float v1 = fmaf(-2.f, c11[j], e1);
            if (v0 < best[j])     { best[j]     = v0; bidx[j]     = n1; }
            if (v1 < best[8 + j]) { best[8 + j] = v1; bidx[8 + j] = n1; }
        }
    }

    // ---- argmin reduce across the 16 lanes holding each row's columns ----
    #pragma unroll
    for (int m = 1; m <= 8; m <<= 1) {
        #pragma unroll
        for (int j = 0; j < 16; ++j) {
            float ov = __shfl_xor(best[j], m);
            int   oi = __shfl_xor(bidx[j], m);
            bool take = (ov < best[j]) | ((ov == best[j]) & (oi < bidx[j]));
            best[j] = take ? ov : best[j];
            bidx[j] = take ? oi : bidx[j];
        }
    }

    // lanes 0 (rows j+0 / 16+j) and 16 (rows 8+j / 24+j) publish results
    if (ln == 0) {
        float ls = 0.f;
        #pragma unroll
        for (int half = 0; half < 2; ++half) {
            #pragma unroll
            for (int j = 0; j < 8; ++j) {
                int m = half * 16 + hi * 8 + j;     // local row within wave tile
                int g = waveBase + m;
                int bi = bidx[half * 8 + j];
                float bv = best[half * 8 + j];
                out[(size_t)IDX_OFF + g] = (float)bi;
                sBest[wave][m] = bi;
                ls += fmaxf(sXsq[wave][m] + bv, 0.f);   // = sum_d (e - x)^2
            }
        }
        atomicAdd(&sLoss, ls);
    }

    // ---- quant_st == embedding[idx] exactly: gather f32 rows ----
    #pragma unroll 4
    for (int r = 0; r < 32; ++r) {
        int n = sBest[wave][r];
        int g = waveBase + r;
        f32x2 q = *(const f32x2*)(emb + (size_t)n * D_DIM + lane * 2);
        *(f32x2*)(out + (size_t)g * D_DIM + lane * 2) = q;
    }

    __syncthreads();
    if (tid == 0) partials[blockIdx.x] = sLoss;
}

// ---------------- final deterministic loss reduction ---------------------------
__global__ __launch_bounds__(256) void vq_final(const float* __restrict__ partials,
                                                float* __restrict__ out) {
    __shared__ float red[256];
    float s = 0.f;
    for (int i = threadIdx.x; i < NBLOCKS; i += 256) s += partials[i];
    red[threadIdx.x] = s;
    __syncthreads();
    for (int st = 128; st > 0; st >>= 1) {
        if (threadIdx.x < st) red[threadIdx.x] += red[threadIdx.x + st];
        __syncthreads();
    }
    // loss = q_latent + 0.25 * e_latent = 1.25 * mean((quant-flat)^2)
    if (threadIdx.x == 0) out[LOSS_OFF] = red[0] * (1.25f / (float)QUANT_ELEMS);
}

extern "C" void kernel_launch(void* const* d_in, const int* in_sizes, int n_in,
                              void* d_out, int out_size, void* d_ws, size_t ws_size,
                              hipStream_t stream) {
    (void)in_sizes; (void)n_in; (void)out_size; (void)ws_size;
    const float* x   = (const float*)d_in[0];     // [32,4096,64] f32
    const float* emb = (const float*)d_in[1];     // [1024,64] f32
    float* out = (float*)d_out;

    unsigned short* wsE  = (unsigned short*)d_ws;                       // 128 KB
    float* wsEsq         = (float*)((char*)d_ws + K_CODES * D_DIM * 2); // 4 KB
    float* partials      = (float*)((char*)d_ws + K_CODES * D_DIM * 2 + K_CODES * 4);

    vq_prep <<<4,       256, 0, stream>>>(emb, wsE, wsEsq);
    vq_main <<<NBLOCKS, 256, 0, stream>>>(x, emb, wsE, wsEsq, out, partials);
    vq_final<<<1,       256, 0, stream>>>(partials, out);
}
